// SelfAttention_37701222924667
// MI455X (gfx1250) — matrix operand order repaired and checked by reading, not compiled, not run on hardware
//
#include <hip/hip_runtime.h>
#include <hip/hip_bf16.h>
#include <stdint.h>

// ---------------------------------------------------------------------------
// Self-attention (B=512, T=128, E=1024, D=1024) for gfx1250 (MI455X).
//
// Phase 0: convert W{q,k,v} f32 -> bf16 once (6 MB, L2-resident).
// Phase 1: QKV projections. Each block keeps a 64-row slab of x resident in
//          LDS as bf16 (x read from HBM exactly ONCE). A-fragments from LDS,
//          B-fragments loaded directly from bf16 weights (L2/WGP$ resident);
//          NO barriers / LDS staging in the main loop.
//          Q/K stored row-major bf16, V stored transposed [B,D,T] bf16.
// Phase 2: fused scores -> causal softmax -> P*V, one workgroup per batch,
//          triangular k-tile skipping, output directly in the reference's
//          transposed [B,D,T] f32 layout (contiguous float4 stores).
// ---------------------------------------------------------------------------

#define BATCH 512
#define SEQ   128
#define EDIM  1024
#define DDIM  1024

typedef __attribute__((ext_vector_type(16))) __bf16 v16bf;
typedef __attribute__((ext_vector_type(8)))  __bf16 v8bf;
typedef __attribute__((ext_vector_type(4)))  __bf16 v4bf;
typedef __attribute__((ext_vector_type(8)))  float  v8f;

union V16U { v16bf v; v8bf h[2]; };

static __device__ inline v16bf join16(v8bf lo, v8bf hi) {
  V16U u; u.h[0] = lo; u.h[1] = hi; return u.v;
}

static __device__ inline v8f wmma_bf16(v16bf a, v16bf b, v8f c) {
  return __builtin_amdgcn_wmma_f32_16x16x32_bf16(
      /*neg_a=*/false, a, /*neg_b=*/false, b,
      /*c_mod=*/(short)0, c, /*reuse_a=*/false, /*reuse_b=*/false);
}

static __device__ inline v4bf cvt4(float4 f) {
  v4bf r = { (__bf16)f.x, (__bf16)f.y, (__bf16)f.z, (__bf16)f.w };
  return r;
}

// ---------------------------------------------------------------------------
// Phase 0: W f32 -> bf16. grid = (D*E/1024, 3), 256 threads, 4 elems/thread.
// ---------------------------------------------------------------------------
__global__ __launch_bounds__(256)
void w_to_bf16_kernel(const float* __restrict__ Wq,
                      const float* __restrict__ Wk,
                      const float* __restrict__ Wv,
                      __bf16* __restrict__ Wb) {
  const int z = blockIdx.y;
  const float* __restrict__ src = (z == 0) ? Wq : ((z == 1) ? Wk : Wv);
  const size_t i = ((size_t)blockIdx.x * 256 + threadIdx.x) * 4;
  const float4 f = *(const float4*)(src + i);
  *(v4bf*)(Wb + (size_t)z * DDIM * EDIM + i) = cvt4(f);
}

// ---------------------------------------------------------------------------
// Phase 1: grid.x = (B*T)/64 = 1024 blocks, 256 threads = 8 waves.
// Waves: 4 along M (4x16 = 64 rows) x 2 along N (2x64 = 128-col n-block).
// ---------------------------------------------------------------------------
#define XSTR 1032  // 1024 + 8 pad: row stride 2064 B (16B-aligned, bank-skewed)

__global__ __launch_bounds__(256)
void qkv_rowslab_kernel(const float* __restrict__ x,
                        const __bf16* __restrict__ Wb,
                        __bf16* __restrict__ Q,
                        __bf16* __restrict__ K,
                        __bf16* __restrict__ Vt) {
  __shared__ __bf16 Xs[64 * XSTR];  // 129 KB: 64 rows of x, full K=1024

  const int tid  = threadIdx.x;
  const int lane = tid & 31;
  const int wave = tid >> 5;
  const int wm   = wave & 3;   // 0..3 along M
  const int wn   = wave >> 2;  // 0..1 along N
  const int l16  = lane & 15;
  const int lhi  = lane >> 4;

  const int mBase = blockIdx.x * 64;
  const int bb    = mBase >> 7;          // batch index (64 | 128)
  const int tBase = (mBase & 127) + wm * 16 + lhi * 8;

  // ---- stage the whole 64x1024 x-slab as bf16 (one-time; x read ONCE) ----
#pragma unroll 4
  for (int j = 0; j < 64; ++j) {
    const int idx = tid + j * 256;          // 0..16383 float4's
    const int row = idx >> 8;               // 256 float4 per row
    const int c4  = (idx & 255) * 4;
    const float4 a = *(const float4*)(x + (size_t)(mBase + row) * EDIM + c4);
    *(v4bf*)&Xs[row * XSTR + c4] = cvt4(a);
  }
  __syncthreads();

  const int arow = (wm * 16 + l16) * XSTR;  // A-fragment row base in Xs

  for (int z = 0; z < 3; ++z) {
    const __bf16* __restrict__ Wz = Wb + (size_t)z * DDIM * EDIM;

    for (int nb = 0; nb < 8; ++nb) {
      const int nBase = nb * 128;

      v8f acc[4];
#pragma unroll
      for (int nt = 0; nt < 4; ++nt)
#pragma unroll
        for (int r = 0; r < 8; ++r) acc[nt][r] = 0.0f;

      // per-lane weight-row pointers (col = nBase + wn*64 + nt*16 + l16)
      const __bf16* wr0 = Wz + (size_t)(nBase + wn * 64 + l16) * EDIM + lhi * 8;

#pragma unroll 2
      for (int kki = 0; kki < 32; ++kki) {
        const int kk = kki * 32;

        // A fragment (ISA 16-bit A layout: two 16B chunks per lane)
        const v16bf a = join16(*(const v8bf*)&Xs[arow + kk + lhi * 8],
                               *(const v8bf*)&Xs[arow + kk + 16 + lhi * 8]);
        // B fragments straight from bf16 weights (same k layout as A: two 16B chunks per lane)
#pragma unroll
        for (int nt = 0; nt < 4; ++nt) {
          const __bf16* wr = wr0 + (size_t)(nt * 16) * EDIM + kk;
          const v16bf bf = join16(*(const v8bf*)wr, *(const v8bf*)(wr + 16));
          acc[nt] = wmma_bf16(a, bf, acc[nt]);
        }
      }

      // ---- store (branch hoisted; constant offsets fold into ioffset) ----
      const int colBase = nBase + wn * 64 + l16;
      if (z == 2) {
        __bf16* p0 = Vt + ((size_t)bb * DDIM + colBase) * SEQ + tBase;
#pragma unroll
        for (int nt = 0; nt < 4; ++nt)
#pragma unroll
          for (int r = 0; r < 8; ++r)
            p0[nt * 16 * SEQ + r] = (__bf16)acc[nt][r];
      } else {
        __bf16* dst = (z == 0) ? Q : K;
        __bf16* p0 = dst + (size_t)(mBase + wm * 16 + lhi * 8) * DDIM + colBase;
#pragma unroll
        for (int nt = 0; nt < 4; ++nt)
#pragma unroll
          for (int r = 0; r < 8; ++r)
            p0[(size_t)r * DDIM + nt * 16] = (__bf16)acc[nt][r];
      }
    }
  }
}

// ---------------------------------------------------------------------------
// Phase 2: per-batch fused attention. Block = 256 threads = 8 waves,
// wave w owns q-rows [16w, 16w+16). Triangular tile skip on scores.
// ---------------------------------------------------------------------------
__global__ __launch_bounds__(256)
void attn_fused_kernel(const __bf16* __restrict__ Q,
                       const __bf16* __restrict__ K,
                       const __bf16* __restrict__ Vt,
                       float* __restrict__ out) {
  const int b    = blockIdx.x;
  const int tid  = threadIdx.x;
  const int lane = tid & 31;
  const int wave = tid >> 5;
  const int l16  = lane & 15;
  const int lhi  = lane >> 4;
  const int q0   = wave * 16;

  __shared__ __bf16 P[8][16][136];  // per-wave softmax(P), row-major over k

  const __bf16* __restrict__ Qb = Q  + (size_t)b * SEQ * DDIM;
  const __bf16* __restrict__ Kb = K  + (size_t)b * SEQ * DDIM;
  const __bf16* __restrict__ Vb = Vt + (size_t)b * DDIM * SEQ;

  // ---------------- scores: S[16 x 128] = Q_tile * K^T -----------------
  v8f s[8];
#pragma unroll
  for (int nt = 0; nt < 8; ++nt)
#pragma unroll
    for (int r = 0; r < 8; ++r) s[nt][r] = 0.0f;

  const int ntmax = wave;  // tiles with nt > wave are fully causally masked
  const __bf16* qrow = Qb + (size_t)(q0 + l16) * DDIM;

  for (int kk = 0; kk < DDIM; kk += 32) {
    const v16bf a = join16(*(const v8bf*)(qrow + kk + lhi * 8),
                           *(const v8bf*)(qrow + kk + 16 + lhi * 8));
    for (int nt = 0; nt <= ntmax; ++nt) {
      const __bf16* kr = Kb + (size_t)(nt * 16 + l16) * DDIM + kk + lhi * 8;
      const v16bf bf = join16(*(const v8bf*)kr, *(const v8bf*)(kr + 16));
      s[nt] = wmma_bf16(a, bf, s[nt]);
    }
  }

  // ---------------- causal mask + scale + row softmax ------------------
  const float scale = 0.03125f;  // 1/sqrt(1024)
  float mx[8], sm[8];
#pragma unroll
  for (int r = 0; r < 8; ++r) mx[r] = -3.0e38f;

  for (int nt = 0; nt <= ntmax; ++nt) {
    const int n = nt * 16 + l16;
#pragma unroll
    for (int r = 0; r < 8; ++r) {
      const int qr = q0 + lhi * 8 + r;
      float v = s[nt][r] * scale;
      if (n > qr) v = -3.0e38f;
      s[nt][r] = v;
      mx[r] = fmaxf(mx[r], v);
    }
  }
#pragma unroll
  for (int r = 0; r < 8; ++r) {
#pragma unroll
    for (int d = 1; d < 16; d <<= 1)
      mx[r] = fmaxf(mx[r], __shfl_xor(mx[r], d, 32));
    sm[r] = 0.0f;
  }
  for (int nt = 0; nt <= ntmax; ++nt) {
#pragma unroll
    for (int r = 0; r < 8; ++r) {
      const float p = __expf(s[nt][r] - mx[r]);
      s[nt][r] = p;
      sm[r] += p;
    }
  }
#pragma unroll
  for (int r = 0; r < 8; ++r) {
#pragma unroll
    for (int d = 1; d < 16; d <<= 1)
      sm[r] += __shfl_xor(sm[r], d, 32);
    sm[r] = 1.0f / sm[r];
  }

  // --------------- write P (bf16) to per-wave LDS ----------------------
#pragma unroll
  for (int nt = 0; nt < 8; ++nt) {
#pragma unroll
    for (int r = 0; r < 8; ++r) {
      const __bf16 val = (nt <= ntmax) ? (__bf16)(s[nt][r] * sm[r]) : (__bf16)0.0f;
      P[wave][lhi * 8 + r][nt * 16 + l16] = val;
    }
  }
  // Only this wave reads its P region: intra-wave DS ordering suffices.

  // --------------- preload P as A-fragments (K = 128 -> 4 steps) -------
  v16bf ap[4];
#pragma unroll
  for (int ks = 0; ks < 4; ++ks) {
    const __bf16* pr = &P[wave][l16][ks * 32];
    ap[ks] = join16(*(const v8bf*)(pr + lhi * 8),
                    *(const v8bf*)(pr + 16 + lhi * 8));
  }

  // --------------- out_tile = P * V, stored as [B, D, T] ---------------
  float* __restrict__ outb = out + (size_t)b * DDIM * SEQ;
  for (int dg = 0; dg < 16; ++dg) {  // 16 groups of 4 d-tiles = 1024 cols
    v8f o[4];
#pragma unroll
    for (int dt = 0; dt < 4; ++dt)
#pragma unroll
      for (int r = 0; r < 8; ++r) o[dt][r] = 0.0f;

#pragma unroll
    for (int ks = 0; ks < 4; ++ks) {
#pragma unroll
      for (int dt = 0; dt < 4; ++dt) {
        const int d = dg * 64 + dt * 16 + l16;
        const __bf16* vr = Vb + (size_t)d * SEQ + ks * 32 + lhi * 8;
        const v16bf bf = join16(*(const v8bf*)vr, *(const v8bf*)(vr + 16));
        o[dt] = wmma_bf16(ap[ks], bf, o[dt]);
      }
    }

#pragma unroll
    for (int dt = 0; dt < 4; ++dt) {
      const int d = dg * 64 + dt * 16 + l16;
      float* po = outb + (size_t)d * SEQ + q0 + lhi * 8;  // 8 consecutive t's
      float4 v0 = { o[dt][0], o[dt][1], o[dt][2], o[dt][3] };
      float4 v1 = { o[dt][4], o[dt][5], o[dt][6], o[dt][7] };
      *(float4*)(po)     = v0;
      *(float4*)(po + 4) = v1;
    }
  }
}

// ---------------------------------------------------------------------------
// Harness entry point.
// ---------------------------------------------------------------------------
extern "C" void kernel_launch(void* const* d_in, const int* in_sizes, int n_in,
                              void* d_out, int out_size, void* d_ws, size_t ws_size,
                              hipStream_t stream) {
  const float* x  = (const float*)d_in[0];
  const float* Wq = (const float*)d_in[1];
  const float* Wk = (const float*)d_in[2];
  const float* Wv = (const float*)d_in[3];
  float* out = (float*)d_out;

  // Workspace: Q, K (row-major [B*T, D]), V^T ([B, D, T]), W bf16 [3][D][E].
  const size_t elems = (size_t)BATCH * SEQ * DDIM;  // 64 Mi elements each
  __bf16* Q  = (__bf16*)d_ws;
  __bf16* K  = Q + elems;
  __bf16* Vt = K + elems;
  __bf16* Wb = Vt + elems;

  dim3 g0((DDIM * EDIM) / 1024, 3);
  w_to_bf16_kernel<<<g0, 256, 0, stream>>>(Wq, Wk, Wv, Wb);

  dim3 g1((BATCH * SEQ) / 64);  // 1024 row-slab blocks
  qkv_rowslab_kernel<<<g1, 256, 0, stream>>>(x, Wb, Q, K, Vt);

  attn_fused_kernel<<<dim3(BATCH), 256, 0, stream>>>(Q, K, Vt, out);
}
